// VQ_Space_11802570130336
// MI455X (gfx1250) — matrix operand order repaired and checked by reading, not compiled, not run on hardware
//
#include <hip/hip_runtime.h>

// ---- problem constants (match reference) ----
#define BB 16384
#define SS 8
#define DD 512
#define KK 2048
#define MT 128          // batch rows per workgroup (8 waves x 16-row M tiles)
#define NT 16           // codes per N tile
#define ROWP 520        // padded LDS row length in bf16 elems (1040B, 16B-aligned, bank-skewed)

typedef __attribute__((ext_vector_type(16))) __bf16 v16bf;
typedef __attribute__((ext_vector_type(8)))  __bf16 v8bf;
typedef __attribute__((ext_vector_type(8)))  float  v8f;

static __device__ inline v8f wmma_bf16(v16bf a, v16bf b, v8f c) {
    // (neg_a, A, neg_b, B, c_mod, C, reuse_a, reuse_b)
    return __builtin_amdgcn_wmma_f32_16x16x32_bf16(false, a, false, b, (short)0, c, false, false);
}

// -------- Pass 1: e2[s,k] = sum_d codebook[s,k,d]^2 (fp32 exact) --------
__global__ __launch_bounds__(256) void vq_e2_kernel(const float* __restrict__ cbg,
                                                    float* __restrict__ e2) {
    const int row  = blockIdx.x * 8 + (threadIdx.x >> 5);   // row in [0, S*K)
    const int lane = threadIdx.x & 31;
    const float* p = cbg + (size_t)row * DD;
    float s = 0.f;
    #pragma unroll 4
    for (int d = lane; d < DD; d += 32) {
        float v = p[d];
        s = fmaf(v, v, s);
    }
    #pragma unroll
    for (int m = 1; m < 32; m <<= 1) s += __shfl_xor(s, m, 32);
    if (lane == 0) e2[row] = s;
}

// -------- Pass 2: distances via bf16 WMMA, argmin, gather, partial loss --------
__global__ __launch_bounds__(256) void vq_main_kernel(const float* __restrict__ xg,
                                                      const float* __restrict__ cbg,
                                                      const float* __restrict__ e2g,
                                                      float* __restrict__ zg,
                                                      float* __restrict__ partial) {
    __shared__ __bf16 xbf[MT * ROWP];    // x tile, bf16, padded rows (~130KB)
    __shared__ __bf16 cbbf[NT * ROWP];   // current codebook N-tile, bf16 (~16KB)
    __shared__ int    idxS[MT];
    __shared__ float  wl[8];

    const int tid  = threadIdx.x;
    const int lane = tid & 31;
    const int wave = tid >> 5;
    const int s    = blockIdx.y;          // slot
    const int b0   = blockIdx.x * MT;     // first batch row of this WG

    // ---- stage x tile (fp32 -> bf16) into fragment-friendly LDS ----
    for (int i = tid; i < MT * DD; i += 256) {
        const int row = i >> 9, d = i & (DD - 1);
        const float v = xg[((b0 + row) * SS + s) * DD + d];
        xbf[row * ROWP + d] = (__bf16)v;
    }

    const int m15  = lane & 15;           // column within tile / A-row id
    const int h    = lane >> 4;           // lane half
    const int arow = wave * 16 + m15;     // this lane's x row (A matrix M index)

    // A-fragment base: 16x32 layout -> one contiguous 16-elem run per lane
    const __bf16* ap = &xbf[arow * ROWP + h * 16];
    // B-fragment base: 32x16 layout -> one contiguous 16-elem run per lane
    const __bf16* bp = &cbbf[m15 * ROWP + h * 16];

    // staging assignment: each thread owns 32 contiguous d of one code row
    const int sn     = tid >> 4;          // code row within tile (0..15)
    const int sdbase = (tid & 15) * 32;   // d base (0..480)
    const float* cbslot = cbg + (size_t)s * KK * DD;

    // register prefetch buffer: 8 x float4 = 32 floats (tile nt's slice)
    float4 rg[8];
    {
        const float4* src = (const float4*)(cbslot + (size_t)sn * DD + sdbase);
        #pragma unroll
        for (int j = 0; j < 8; ++j) rg[j] = src[j];
    }

    float bestD[8];
    int   bestI[8];
    #pragma unroll
    for (int r = 0; r < 8; ++r) { bestD[r] = 3.4e38f; bestI[r] = 0; }

    for (int nt = 0; nt < KK / NT; ++nt) {
        __syncthreads();   // previous compute done reading cbbf (also covers x staging)
        // ---- convert + store prefetched tile nt into LDS (cheap: 4 x ds_store_b128) ----
        #pragma unroll
        for (int j = 0; j < 8; j += 2) {
            const float4 a = rg[j], b = rg[j + 1];
            v8bf w;
            w[0] = (__bf16)a.x; w[1] = (__bf16)a.y; w[2] = (__bf16)a.z; w[3] = (__bf16)a.w;
            w[4] = (__bf16)b.x; w[5] = (__bf16)b.y; w[6] = (__bf16)b.z; w[7] = (__bf16)b.w;
            *(v8bf*)(&cbbf[sn * ROWP + sdbase + j * 4]) = w;
        }
        __syncthreads();

        // ---- issue global loads for tile nt+1 now; they complete during compute ----
        if (nt + 1 < KK / NT) {
            const float4* src =
                (const float4*)(cbslot + ((size_t)(nt + 1) * NT + sn) * DD + sdbase);
            #pragma unroll
            for (int j = 0; j < 8; ++j) rg[j] = src[j];
        }

        v8f acc0 = {0.f,0.f,0.f,0.f,0.f,0.f,0.f,0.f};
        v8f acc1 = {0.f,0.f,0.f,0.f,0.f,0.f,0.f,0.f};
        #pragma unroll
        for (int c = 0; c < DD / 32; ++c) {
            v8bf a0 = *(const v8bf*)(ap + c * 32);
            v8bf a1 = *(const v8bf*)(ap + c * 32 + 8);
            v8bf bl = *(const v8bf*)(bp + c * 32);
            v8bf bh = *(const v8bf*)(bp + c * 32 + 8);
            v16bf A  = __builtin_shufflevector(a0, a1, 0,1,2,3,4,5,6,7,8,9,10,11,12,13,14,15);
            v16bf Bm = __builtin_shufflevector(bl, bh, 0,1,2,3,4,5,6,7,8,9,10,11,12,13,14,15);
            if (c & 1) acc1 = wmma_bf16(A, Bm, acc1);
            else       acc0 = wmma_bf16(A, Bm, acc0);
        }
        v8f acc = acc0 + acc1;

        // dist = ||e||^2 - 2 x.e  (per-row ||x||^2 is argmin-invariant)
        const float e2v = e2g[s * KK + nt * NT + m15];
        const int   ki  = nt * NT + m15;
        #pragma unroll
        for (int r = 0; r < 8; ++r) {
            const float dist = fmaf(-2.f, acc[r], e2v);
            if (dist < bestD[r]) { bestD[r] = dist; bestI[r] = ki; }
        }
    }

    // ---- argmin across the 16 lanes of each half (C layout: row = r + 8*h) ----
    #pragma unroll
    for (int r = 0; r < 8; ++r) {
        float bd = bestD[r];
        int   bi = bestI[r];
        #pragma unroll
        for (int m = 1; m <= 8; m <<= 1) {       // masks 1,2,4,8 stay inside each half
            const float od = __shfl_xor(bd, m, 32);
            const int   oi = __shfl_xor(bi, m, 32);
            if (od < bd || (od == bd && oi < bi)) { bd = od; bi = oi; }
        }
        if (m15 == 0) idxS[wave * 16 + h * 8 + r] = bi;
    }
    __syncthreads();

    // ---- gather q (exact fp32), write z = q, accumulate commitment loss ----
    float lsum = 0.f;
    for (int i = tid; i < MT * DD; i += 256) {
        const int row = i >> 9, d = i & (DD - 1);
        const int k   = idxS[row];
        const float q = cbg[((size_t)s * KK + k) * DD + d];
        const int  gi = ((b0 + row) * SS + s) * DD + d;
        const float xv = xg[gi];
        zg[gi] = q;
        const float df = q - xv;
        lsum = fmaf(df, df, lsum);
    }
    #pragma unroll
    for (int m = 1; m < 32; m <<= 1) lsum += __shfl_xor(lsum, m, 32);
    if (lane == 0) wl[wave] = lsum;
    __syncthreads();
    if (tid == 0) {
        float t = 0.f;
        for (int w = 0; w < 8; ++w) t += wl[w];   // fixed order -> deterministic
        partial[blockIdx.y * gridDim.x + blockIdx.x] = t;
    }
}

// -------- Pass 3: deterministic fixed-order reduction of 1024 partials --------
__global__ __launch_bounds__(256) void vq_loss_reduce(const float* __restrict__ partial,
                                                      float* __restrict__ out) {
    __shared__ float red[256];
    const int t = threadIdx.x;
    float s = partial[t] + partial[t + 256] + partial[t + 512] + partial[t + 768];
    red[t] = s;
    __syncthreads();
    for (int off = 128; off > 0; off >>= 1) {
        if (t < off) red[t] += red[t + off];
        __syncthreads();
    }
    if (t == 0) out[0] = red[0] * (1.0f / 67108864.0f);   // / (B*S*D), exact pow2
}

extern "C" void kernel_launch(void* const* d_in, const int* in_sizes, int n_in,
                              void* d_out, int out_size, void* d_ws, size_t ws_size,
                              hipStream_t stream) {
    const float* xg  = (const float*)d_in[0];   // (B,S,D) fp32
    const float* cbg = (const float*)d_in[1];   // (S,K,D) fp32
    float* zg   = (float*)d_out;                // z, then scalar loss at the end
    float* loss = zg + (out_size - 1);

    float* e2      = (float*)d_ws;              // S*K floats  (64 KB)
    float* partial = e2 + SS * KK;              // 1024 floats

    vq_e2_kernel<<<(SS * KK) / 8, 256, 0, stream>>>(cbg, e2);
    vq_main_kernel<<<dim3(BB / MT, SS), 256, 0, stream>>>(xg, cbg, e2, zg, partial);
    vq_loss_reduce<<<1, 256, 0, stream>>>(partial, loss);
}
